// PC_forecasting_model_0_0_5454608466691
// MI455X (gfx1250) — compile-verified, matrix operand order checked by reading, not run on hardware
//
#include <hip/hip_runtime.h>
#include <cstdint>
#include <cstddef>

// ---------------- WMMA plumbing (gfx1250, wave32) ----------------
typedef __attribute__((ext_vector_type(16))) __bf16 v16bf;
typedef __attribute__((ext_vector_type(8)))  float  v8f;

__device__ __forceinline__ uint16_t f2bf(float f) {
  uint32_t u = __float_as_uint(f);
  uint32_t r = u + 0x7FFFu + ((u >> 16) & 1u);   // round-to-nearest-even
  return (uint16_t)(r >> 16);
}

// Load one 16x32 bf16 operand fragment from a row-major [rows x ld] bf16
// (uint16 storage) array. Per ISA 7.12.2: lane&15 = M row; lanes 0-15 hold
// K=0..7 & 16..23, lanes 16-31 hold K=8..15 & 24..31, packed 2/dword.
__device__ __forceinline__ v16bf load_frag(const uint16_t* base, int ld) {
  int lane = threadIdx.x & 31;
  int m    = lane & 15;
  int kh   = (lane >> 4) << 2;            // dword offset: 0 or 4
  const uint32_t* row = (const uint32_t*)(base + (size_t)m * ld);
  union { v16bf v; uint32_t u[8]; } f;
#pragma unroll
  for (int j = 0; j < 4; ++j) {
    f.u[j]     = row[kh + j];             // K 0..7   (or 8..15)
    f.u[4 + j] = row[kh + 8 + j];         // K 16..23 (or 24..31)
  }
  return f.v;
}

__device__ __forceinline__ v8f wmma_bf16(v16bf a, v16bf b, v8f c) {
  return __builtin_amdgcn_wmma_f32_16x16x32_bf16(false, a, false, b,
                                                 (short)0, c, false, false);
}

__device__ __forceinline__ float leaky(float x) { return x >= 0.f ? x : 0.2f * x; }

// ---- async global->LDS copy (CDNA5, ASYNCcnt-tracked), 16B per lane ----
__device__ __forceinline__ void async_ld_b128(uint32_t lds_off, uint64_t gaddr) {
  asm volatile("global_load_async_to_lds_b128 %0, %1, off"
               :: "v"(lds_off), "v"(gaddr) : "memory");
}
// stage one contiguous 16KB chunk (64 rows x 256B) with 4 x b128 per thread
__device__ __forceinline__ void stage_chunk(uint16_t* lds_dst, const uint16_t* gsrc,
                                            int tid) {
  uint32_t l = (uint32_t)(uintptr_t)lds_dst + (uint32_t)tid * 16u;
  uint64_t g = (uint64_t)(uintptr_t)gsrc + (uint64_t)tid * 16u;
#pragma unroll
  for (int p = 0; p < 4; ++p) async_ld_b128(l + p * 4096u, g + p * 4096u);
}

// Problem constants (from reference setup_inputs)
#define TT 5
#define NB 4
#define MM 4096
#define CC 128
#define GG 1024
#define NH 4
#define NTOT 16384
#define KNN 16
#define HC 512

// ---------------- zero init (d_ws is poisoned) ----------------
__global__ __launch_bounds__(256) void k_zero(float* __restrict__ a, int n) {
  int t = blockIdx.x * 256 + threadIdx.x;
  if (t < n) a[t] = 0.f;
}

// ---------------- pc: f32 -> bf16 + row squared norms ----------------
__global__ __launch_bounds__(256) void k_prep_pc(const float* __restrict__ pc,
                                                 uint16_t* __restrict__ pcb,
                                                 float* __restrict__ norms) {
  int wave = threadIdx.x >> 5, lane = threadIdx.x & 31;
  int row = blockIdx.x * 8 + wave;                    // 81920 rows total
  const float4 v = *(const float4*)(pc + (size_t)row * CC + lane * 4);
  uint32_t p0 = (uint32_t)f2bf(v.x) | ((uint32_t)f2bf(v.y) << 16);
  uint32_t p1 = (uint32_t)f2bf(v.z) | ((uint32_t)f2bf(v.w) << 16);
  uint32_t* dst = (uint32_t*)(pcb + (size_t)row * CC) + lane * 2;
  dst[0] = p0; dst[1] = p1;
  float nrm = v.x * v.x + v.y * v.y + v.z * v.z + v.w * v.w;
#pragma unroll
  for (int off = 16; off; off >>= 1) nrm += __shfl_xor(nrm, off, 32);
  if (lane == 0) norms[row] = nrm;
}

// ---------------- weight transpose + bf16: dst[n*K+k] = src[k*N+n] ----------------
__global__ __launch_bounds__(256) void k_transpose_bf(const float* __restrict__ src,
                                                      uint16_t* __restrict__ dst,
                                                      int K, int N) {
  int tid = blockIdx.x * 256 + threadIdx.x;
  if (tid >= K * N) return;
  int n = tid / K, k = tid - n * K;
  dst[(size_t)n * K + k] = f2bf(src[(size_t)k * N + n]);
}

// ---------------- fused 2-layer MLP + mean pool -> gf_sum[t][b][g] ----------------
// grid = T*B*16 blocks, 256 threads (8 waves); each block streams 256 rows.
__global__ __launch_bounds__(256) void k_mlp_pool(
    const uint16_t* __restrict__ pcb, const uint16_t* __restrict__ W1t,
    const float* __restrict__ b1, const uint16_t* __restrict__ W2t,
    const float* __restrict__ b2, float* __restrict__ gf_sum) {
  __shared__ __align__(16) uint16_t h1[16 * CC];
  __shared__ float colsum[GG];
  int tid = threadIdx.x;
  int wave = tid >> 5, lane = tid & 31;
  int wg = blockIdx.x;
  int t = wg >> 6, b = (wg >> 4) & 3, chunk = wg & 15;
  for (int n = tid; n < GG; n += 256) colsum[n] = 0.f;
  __syncthreads();
  // W1 fragments are invariant: hoist explicitly (32 VGPRs)
  v16bf w1f[4];
#pragma unroll
  for (int q = 0; q < 4; ++q)
    w1f[q] = load_frag(W1t + (size_t)(wave * 16) * CC + q * 32, CC);
#pragma unroll 1
  for (int it = 0; it < 16; ++it) {
    size_t rowbase = (size_t)t * NTOT + (size_t)b * MM + chunk * 256 + it * 16;
    // stage 1: [16x128] @ W1 -> bf16 tile in LDS (wave w owns cols w*16..)
    v8f acc = {};
#pragma unroll
    for (int q = 0; q < 4; ++q) {
      v16bf a = load_frag(pcb + rowbase * CC + q * 32, CC);
      acc = wmma_bf16(a, w1f[q], acc);
    }
    {
      int n = wave * 16 + (lane & 15);
      float bias = b1[n];
#pragma unroll
      for (int r = 0; r < 8; ++r) {
        int m = r + ((lane >> 4) << 3);
        h1[m * CC + n] = f2bf(leaky(acc[r] + bias));
      }
    }
    __syncthreads();
    // h1 fragments for this iteration: hoist explicitly (32 VGPRs)
    v16bf h1f[4];
#pragma unroll
    for (int q = 0; q < 4; ++q) h1f[q] = load_frag(h1 + q * 32, CC);
    // stage 2: [16x128] @ W2 -> column sums (wave w owns cols w*128..)
    // keep rolled: W2 frags are re-read from L2 instead of spilling 256 VGPRs
#pragma unroll 1
    for (int sub = 0; sub < 8; ++sub) {
      int gcol = (wave * 8 + sub) * 16;
      v8f acc2 = {};
#pragma unroll
      for (int q = 0; q < 4; ++q) {
        v16bf w = load_frag(W2t + (size_t)gcol * CC + q * 32, CC);
        acc2 = wmma_bf16(h1f[q], w, acc2);
      }
      int n = gcol + (lane & 15);
      float bias = b2[n];
      float s = 0.f;
#pragma unroll
      for (int r = 0; r < 8; ++r) s += leaky(acc2[r] + bias);
      atomicAdd(&colsum[n], s);
    }
    __syncthreads();
  }
  for (int n = tid; n < GG; n += 256)
    atomicAdd(&gf_sum[((size_t)t * NB + b) * GG + n], colsum[n]);
}

// ---------------- temporal attention: attn[b][t] ----------------
__global__ __launch_bounds__(64) void k_attn(const float* __restrict__ gf_sum,
                                             float* __restrict__ attn) {
  __shared__ float sc[16];
  int tid = threadIdx.x;
  if (tid < 16) {
    int b = tid >> 2, t = tid & 3;
    const float* q  = gf_sum + ((TT - 1) * NB + b) * GG;
    const float* kv = gf_sum + (t * NB + b) * GG;
    float d = 0.f;
    for (int g = 0; g < GG; ++g) d += q[g] * kv[g];
    // gf = sum/M; score = q.kv/sqrt(G)  ->  d / (M*M*32)
    sc[tid] = d / ((float)MM * (float)MM * 32.f);
  }
  __syncthreads();
  if (tid < 4) {
    int b = tid;
    float mx = sc[b * 4];
    for (int t = 1; t < 4; ++t) mx = fmaxf(mx, sc[b * 4 + t]);
    float e[4], s = 0.f;
    for (int t = 0; t < 4; ++t) { e[t] = __expf(sc[b * 4 + t] - mx); s += e[t]; }
    for (int t = 0; t < 4; ++t) attn[b * 4 + t] = e[t] / s;
  }
}

// ---------------- C = A[M x 128] * Bt[N x 128]^T (both bf16, f32 out) ----------------
__global__ __launch_bounds__(256) void k_gemm_nt(const uint16_t* __restrict__ A,
                                                 const uint16_t* __restrict__ Bt,
                                                 float* __restrict__ Cmat,
                                                 int Mrows, int Ncols) {
  int wave = threadIdx.x >> 5, lane = threadIdx.x & 31;
  int ntn = Ncols >> 4;
  int tile = blockIdx.x * 8 + wave;
  if (tile >= (Mrows >> 4) * ntn) return;
  int m0 = (tile / ntn) << 4;
  int n0 = (tile % ntn) << 4;
  v8f acc = {};
#pragma unroll
  for (int q = 0; q < 4; ++q) {
    v16bf a = load_frag(A  + (size_t)m0 * CC + q * 32, CC);
    v16bf w = load_frag(Bt + (size_t)n0 * CC + q * 32, CC);
    acc = wmma_bf16(a, w, acc);
  }
  int n = n0 + (lane & 15);
#pragma unroll
  for (int r = 0; r < 8; ++r) {
    int m = m0 + r + ((lane >> 4) << 3);
    Cmat[(size_t)m * Ncols + n] = acc[r];
  }
}

// ---------------- fused KNN: async-staged WMMA score tile + streaming top-16 --------
// grid (128 row-tiles, B, 4 past frames), 256 threads. 32 query rows / block.
// Candidate chunks (64 rows x 256B, contiguous) are double-buffered into LDS
// with GLOBAL_LOAD_ASYNC_TO_LDS_B128 overlapped against the WMMA of the
// current chunk; ASYNCcnt in-order completion => s_wait_asynccnt 4 after
// issuing the next chunk guarantees the current one landed.
__global__ __launch_bounds__(256) void k_knn(const uint16_t* __restrict__ pcb,
                                             const float* __restrict__ norms,
                                             int* __restrict__ nbr_all) {
  __shared__ __align__(16) uint16_t cand[2][64 * CC];   // 2 x 16KB
  __shared__ float sc[32 * 64];
  __shared__ float mv[32 * 8 * 16];
  __shared__ int   mi[32 * 8 * 16];
  int tid  = threadIdx.x;
  int wave = tid >> 5, lane = tid & 31;
  int rt = blockIdx.x, b = blockIdx.y, i = blockIdx.z;
  size_t qbase = (size_t)(TT - 1) * NTOT + (size_t)b * MM + rt * 32;  // last frame
  size_t cbase = (size_t)i * NTOT + (size_t)b * MM;                   // past frame i
  const float* xn = norms + (size_t)i * NTOT + (size_t)b * MM;
  float hv[KNN]; int hidx[KNN];
#pragma unroll
  for (int s = 0; s < KNN; ++s) { hv[s] = -3.0e38f; hidx[s] = b * MM; }
  int tile_r = (wave & 1) << 4;
  int tile_c = (wave >> 1) << 4;
  int prow  = tid & 31;           // scan role: row
  int pcol0 = (tid >> 5) << 3;    // scan role: 8-col slice
  // query-row fragments: invariant across all chunks, keep in 32 VGPRs
  v16bf afr[4];
#pragma unroll
  for (int q = 0; q < 4; ++q)
    afr[q] = load_frag(pcb + (qbase + tile_r) * CC + q * 32, CC);
  // prefetch chunk 0
  stage_chunk(&cand[0][0], pcb + cbase * CC, tid);
#pragma unroll 1
  for (int ch = 0; ch < 64; ++ch) {
    int cur = ch & 1;
    int col0 = ch * 64;
    if (ch + 1 < 64) {
      stage_chunk(&cand[cur ^ 1][0], pcb + (cbase + (size_t)(ch + 1) * 64) * CC, tid);
      asm volatile("s_wait_asynccnt 0x4" ::: "memory");   // chunk ch landed
    } else {
      asm volatile("s_wait_asynccnt 0x0" ::: "memory");
    }
    __syncthreads();                                      // LDS visible to all waves
    v8f acc = {};
#pragma unroll
    for (int q = 0; q < 4; ++q) {
      v16bf bb = load_frag(&cand[cur][0] + (size_t)tile_c * CC + q * 32, CC);
      acc = wmma_bf16(afr[q], bb, acc);
    }
#pragma unroll
    for (int r = 0; r < 8; ++r) {
      int m = tile_r + r + ((lane >> 4) << 3);
      int n = tile_c + (lane & 15);
      sc[m * 64 + n] = acc[r];
    }
    __syncthreads();
#pragma unroll
    for (int jj = 0; jj < 8; ++jj) {
      int n = pcol0 + jj;
      int col = col0 + n;
      float val = 2.f * sc[prow * 64 + n] - xn[col];   // -dist (query norm const)
      if (val > hv[0]) {                               // sorted-ascending insert
        int j = 1;
        while (j < KNN && val > hv[j]) { hv[j-1] = hv[j]; hidx[j-1] = hidx[j]; ++j; }
        hv[j-1] = val; hidx[j-1] = b * MM + col;
      }
    }
    __syncthreads();                                    // sc + buffer reuse guard
  }
  // merge 8 partial top-16 lists per row
  {
    float* mvp = mv + (prow * 8 + (tid >> 5)) * KNN;
    int*   mip = mi + (prow * 8 + (tid >> 5)) * KNN;
#pragma unroll
    for (int s = 0; s < KNN; ++s) { mvp[s] = hv[s]; mip[s] = hidx[s]; }
  }
  __syncthreads();
  if (tid < 32) {
    int row = tid;
    float* rv = mv + row * 128;
    int*   ri = mi + row * 128;
    int qglob = i * NTOT + b * MM + rt * 32 + row;
    int* out = nbr_all + (size_t)qglob * KNN;
    for (int s = 0; s < KNN; ++s) {
      float best = -3.3e38f; int bj = 0;
      for (int j = 0; j < 128; ++j) {
        float v = rv[j];
        if (v > best) { best = v; bj = j; }
      }
      out[s] = ri[bj];
      rv[bj] = -3.3e38f;
    }
  }
}

// ---------------- GATv2 aggregation, one block per query, wave per head ----------------
__global__ __launch_bounds__(128) void k_gat(const float* __restrict__ xl,
                                             const float* __restrict__ xr,
                                             const int* __restrict__ nbr,
                                             const float* __restrict__ att_w,
                                             const float* __restrict__ attn, int i,
                                             float* __restrict__ weighted) {
  __shared__ float outsum[CC];
  int tid = threadIdx.x;
  int h = tid >> 5, lane = tid & 31;
  int n = blockIdx.x;
  int b = n >> 12;
  float aw = attn[b * 4 + i] * (1.f / NH);
  outsum[tid] = 0.f;
  __syncthreads();
  float xr4[4], aw4[4];
#pragma unroll
  for (int j = 0; j < 4; ++j) {
    int c = lane + 32 * j;
    xr4[j] = xr[(size_t)n * HC + h * CC + c];
    aw4[j] = att_w[h * CC + c];
  }
  float xv[KNN][4], ek[KNN];
#pragma unroll
  for (int k = 0; k < KNN; ++k) {
    int idx = nbr[n * KNN + k];
    const float* xlr = xl + (size_t)idx * HC + h * CC;
    float e = 0.f;
#pragma unroll
    for (int j = 0; j < 4; ++j) {
      float v = xlr[lane + 32 * j];
      xv[k][j] = v;
      e += aw4[j] * leaky(v + xr4[j]);
    }
#pragma unroll
    for (int off = 16; off; off >>= 1) e += __shfl_xor(e, off, 32);
    ek[k] = e;
  }
  float mx = ek[0];
#pragma unroll
  for (int k = 1; k < KNN; ++k) mx = fmaxf(mx, ek[k]);
  float ssum = 0.f;
#pragma unroll
  for (int k = 0; k < KNN; ++k) { ek[k] = __expf(ek[k] - mx); ssum += ek[k]; }
  float inv = 1.f / ssum;
  float o[4] = {0.f, 0.f, 0.f, 0.f};
#pragma unroll
  for (int k = 0; k < KNN; ++k) {
    float a = ek[k] * inv;
#pragma unroll
    for (int j = 0; j < 4; ++j) o[j] += a * xv[k][j];
  }
#pragma unroll
  for (int j = 0; j < 4; ++j) atomicAdd(&outsum[lane + 32 * j], o[j]);
  __syncthreads();
  weighted[(size_t)n * CC + tid] += aw * outsum[tid];
}

// ---------------- concat(last, weighted) ----------------
__global__ __launch_bounds__(256) void k_finalize(const float* __restrict__ pc_last,
                                                  const float* __restrict__ weighted,
                                                  float* __restrict__ out) {
  int tid = blockIdx.x * 256 + threadIdx.x;      // 0 .. N*C
  int n = tid >> 7, c = tid & 127;
  out[(size_t)n * 256 + c]       = pc_last[tid];
  out[(size_t)n * 256 + 128 + c] = weighted[tid];
}

extern "C" void kernel_launch(void* const* d_in, const int* in_sizes, int n_in,
                              void* d_out, int out_size, void* d_ws, size_t ws_size,
                              hipStream_t stream) {
  (void)in_sizes; (void)n_in; (void)out_size; (void)ws_size;
  const float* pc    = (const float*)d_in[0];
  const float* W1    = (const float*)d_in[1];
  const float* b1    = (const float*)d_in[2];
  const float* W2    = (const float*)d_in[3];
  const float* b2    = (const float*)d_in[4];
  const float* Wl    = (const float*)d_in[5];
  const float* Wr    = (const float*)d_in[6];
  const float* att_w = (const float*)d_in[7];
  float* out = (float*)d_out;

  uint8_t* p = (uint8_t*)d_ws;
  auto carve = [&](size_t bytes) -> void* {
    void* r = (void*)p;
    p += (bytes + 255) & ~(size_t)255;
    return r;
  };
  uint16_t* pcb   = (uint16_t*)carve((size_t)TT * NTOT * CC * 2);   // 20 MB
  float*    norms = (float*)   carve((size_t)TT * NTOT * 4);
  uint16_t* W1t   = (uint16_t*)carve((size_t)CC * CC * 2);
  uint16_t* W2t   = (uint16_t*)carve((size_t)GG * CC * 2);
  uint16_t* Wlt   = (uint16_t*)carve((size_t)HC * CC * 2);
  uint16_t* Wrt   = (uint16_t*)carve((size_t)HC * CC * 2);
  float*    gfsum = (float*)   carve((size_t)TT * NB * GG * 4);
  float*    attn  = (float*)   carve(256);
  float*    xl    = (float*)   carve((size_t)NTOT * HC * 4);        // 32 MB
  float*    xr    = (float*)   carve((size_t)NTOT * HC * 4);        // 32 MB
  int*      nbr   = (int*)     carve((size_t)4 * NTOT * KNN * 4);   // 4 MB
  float*    wsum  = (float*)   carve((size_t)NTOT * CC * 4);        // 8 MB

  // re-zero accumulators every call (graph-replay safe, d_ws poisoned once)
  k_zero<<<(TT * NB * GG + 255) / 256, 256, 0, stream>>>(gfsum, TT * NB * GG);
  k_zero<<<(NTOT * CC + 255) / 256, 256, 0, stream>>>(wsum, NTOT * CC);

  k_prep_pc<<<(TT * NTOT) / 8, 256, 0, stream>>>(pc, pcb, norms);
  k_transpose_bf<<<(CC * CC + 255) / 256, 256, 0, stream>>>(W1, W1t, CC, CC);
  k_transpose_bf<<<(CC * GG + 255) / 256, 256, 0, stream>>>(W2, W2t, CC, GG);
  k_transpose_bf<<<(CC * HC + 255) / 256, 256, 0, stream>>>(Wl, Wlt, CC, HC);
  k_transpose_bf<<<(CC * HC + 255) / 256, 256, 0, stream>>>(Wr, Wrt, CC, HC);

  k_mlp_pool<<<TT * NB * 16, 256, 0, stream>>>(pcb, W1t, b1, W2t, b2, gfsum);
  k_attn<<<1, 64, 0, stream>>>(gfsum, attn);

  // xl = last @ Wl : 1024*32 tiles / 8 waves
  k_gemm_nt<<<(NTOT / 16) * (HC / 16) / 8, 256, 0, stream>>>(
      pcb + (size_t)(TT - 1) * NTOT * CC, Wlt, xl, NTOT, HC);

  dim3 kg(MM / 32, NB, 4);
  k_knn<<<kg, 256, 0, stream>>>(pcb, norms, nbr);

  for (int i = 0; i < 4; ++i) {
    k_gemm_nt<<<(NTOT / 16) * (HC / 16) / 8, 256, 0, stream>>>(
        pcb + (size_t)i * NTOT * CC, Wrt, xr, NTOT, HC);
    k_gat<<<NTOT, 128, 0, stream>>>(xl, xr, nbr + (size_t)i * NTOT * KNN,
                                    att_w, attn, i, wsum);
  }
  k_finalize<<<(NTOT * CC) / 256, 256, 0, stream>>>(
      pc + (size_t)(TT - 1) * NTOT * CC, wsum, out);
}
